// TwoDimPlanesModel_50586124812690
// MI455X (gfx1250) — matrix operand order repaired and checked by reading, not compiled, not run on hardware
//
#include <hip/hip_runtime.h>

#define RES      512
#define CCH      48
#define DC       128
#define NHID     3
#define TPB      256
#define TILE_PTS 256

typedef __attribute__((ext_vector_type(16))) _Float16 v16h;
typedef __attribute__((ext_vector_type(8)))  _Float16 v8h;
typedef __attribute__((ext_vector_type(8)))  float    v8f;

// ---------------- LDS layout (units: _Float16 halves) ----------------
// Weight region is reused: density MLP staged first, rgb MLP restaged after a barrier.
#define OFF_W0     0                         // 64 x 128  (K padded 48->64)
#define OFF_WH     (64 * DC)                 // 8192, then NHID * 128x128
#define OFF_HEAD   (OFF_WH + NHID * DC * DC) // 57344, 128 x 16 (N padded)
#define W_HALVES   (OFF_HEAD + DC * 16)      // 59392
#define OFF_FEAT   W_HALVES                  // 256 points x 64 halves
#define OFF_ACT    (OFF_FEAT + TILE_PTS*64)  // 8 waves x (32 x 128) halves
#define OFF_BIAS   (OFF_ACT + 8 * 32 * DC)   // 108544: 544 f32 (b0|bh*3|head16)
#define SMEM_HALVES (OFF_BIAS + 2 * 544)     // 109632 halves
#define SMEM_BYTES  (SMEM_HALVES * 2)        // 219264 B (< 320KB WGP LDS)

// ---------------------------------------------------------------------
// Tri-plane bilinear sampling: one thread = one point, fused over 3 planes,
// f32 accumulate, write f16 row (K padded to 64 with zeros) into LDS.
// Planes total 151MB < 192MB L2, so the gather is L2-resident.
// ---------------------------------------------------------------------
__device__ __forceinline__ void sample_feat(const float* __restrict__ points,
                                            const float* __restrict__ planes,
                                            _Float16* featb, int tid, int pbase)
{
  const int p = pbase + tid;
  const float px = points[(size_t)p * 3 + 0];
  const float py = points[(size_t)p * 3 + 1];
  const float pz = points[(size_t)p * 3 + 2];

  const float* bases[3];
  float w00[3], w01[3], w10[3], w11[3];
#pragma unroll
  for (int d = 0; d < 3; ++d) {
    // PROJ_MATS: d0 -> (y,z), d1 -> (x,z), d2 -> (x,y)
    const float u = (d == 0) ? py : px;
    const float v = (d == 2) ? py : pz;
    const float ix = (u + 1.f) * 0.5f * (float)(RES - 1);
    const float iy = (v + 1.f) * 0.5f * (float)(RES - 1);
    float fx = floorf(ix), fy = floorf(iy);
    fx = fminf(fmaxf(fx, 0.f), (float)(RES - 2));
    fy = fminf(fmaxf(fy, 0.f), (float)(RES - 2));
    const float wx = ix - fx, wy = iy - fy;
    w00[d] = (1.f - wx) * (1.f - wy);
    w01[d] = wx * (1.f - wy);
    w10[d] = (1.f - wx) * wy;
    w11[d] = wx * wy;
    bases[d] = planes + (size_t)d * CCH * RES * RES
                      + (size_t)((int)fy) * RES + (int)fx;
  }

  _Float16* row = featb + (size_t)tid * 64;
  for (int c = 0; c < CCH; ++c) {
    float a = 0.f;
#pragma unroll
    for (int d = 0; d < 3; ++d) {
      const float* q = bases[d] + (size_t)c * (RES * RES);
      a = fmaf(q[0],       w00[d], a);
      a = fmaf(q[1],       w01[d], a);
      a = fmaf(q[RES],     w10[d], a);
      a = fmaf(q[RES + 1], w11[d], a);
    }
    row[c] = (_Float16)a;
  }
#pragma unroll
  for (int c = CCH; c < 64; ++c) row[c] = (_Float16)0.f;
}

// ---------------------------------------------------------------------
// Stage f32 weights -> LDS f16 in WMMA B-fragment order, plus biases -> LDS f32.
// B frag flat = ((nt*KSTEPS + ks)*32 + lane)*16 + h; lane<16 -> K=ks*32+h,
// lane>=16 -> K=ks*32+16+h, N = nt*16 + (lane&15). Compute-time B load is then
// one contiguous 32B per lane (2x ds_load_b128).
// ---------------------------------------------------------------------
__device__ __forceinline__ void stage_weights(_Float16* smem,
                                              const float* __restrict__ w0,
                                              const float* __restrict__ wh,
                                              const float* __restrict__ hw, int hN,
                                              const float* __restrict__ b0,
                                              const float* __restrict__ bh,
                                              const float* __restrict__ hb,
                                              int tid)
{
  _Float16* wbuf = smem;
  // Layer 0: K=64 (48 real), N=128, KSTEPS=2
  for (int i = tid; i < 64 * DC; i += TPB) {
    const int h  = i & 15;
    const int ln = (i >> 4) & 31;
    const int ks = (i >> 9) & 1;
    const int nt = i >> 10;
    const int n  = nt * 16 + (ln & 15);
    const int k  = ks * 32 + ((ln & 16) ? 16 : 0) + h;
    wbuf[OFF_W0 + i] = (_Float16)((k < CCH) ? w0[(size_t)k * DC + n] : 0.f);
  }
  // Hidden layers: K=128, N=128, KSTEPS=4
  for (int l = 0; l < NHID; ++l) {
    const float* g = wh + (size_t)l * DC * DC;
    for (int i = tid; i < DC * DC; i += TPB) {
      const int h  = i & 15;
      const int ln = (i >> 4) & 31;
      const int ks = (i >> 9) & 3;
      const int nt = i >> 11;
      const int n  = nt * 16 + (ln & 15);
      const int k  = ks * 32 + ((ln & 16) ? 16 : 0) + h;
      wbuf[OFF_WH + l * DC * DC + i] = (_Float16)g[(size_t)k * DC + n];
    }
  }
  // Head: K=128, N padded to 16 (hN real columns)
  for (int i = tid; i < DC * 16; i += TPB) {
    const int h  = i & 15;
    const int ln = (i >> 4) & 31;
    const int ks = (i >> 9) & 3;
    const int n  = ln & 15;
    const int k  = ks * 32 + ((ln & 16) ? 16 : 0) + h;
    wbuf[OFF_HEAD + i] = (_Float16)((n < hN) ? hw[(size_t)k * hN + n] : 0.f);
  }
  // Biases -> LDS f32: [0..127]=b0, [128..511]=bh, [512..527]=head (padded)
  float* bl = (float*)(smem + OFF_BIAS);
  for (int i = tid; i < DC; i += TPB)        bl[i]      = b0[i];
  for (int i = tid; i < NHID * DC; i += TPB) bl[DC + i] = bh[i];
  if (tid < 16) bl[4 * DC + tid] = (tid < hN) ? hb[tid] : 0.f;
}

// ---------------------------------------------------------------------
// A-fragment (16x32 f16): lane M = lane&15; halves 0..7 = K k0..k0+7,
// halves 8..15 = K k0+8..k0+15, where k0 = ks*32 + (lane>=16 ? 16 : 0).
// ---------------------------------------------------------------------
__device__ __forceinline__ v16h load_afrag(const _Float16* inrow, int lane, int ks)
{
  const int k0 = ks * 32 + ((lane & 16) ? 16 : 0);
  v8h lo = *(const v8h*)(inrow + k0);
  v8h hi = *(const v8h*)(inrow + k0 + 8);
  v16h a;
#pragma unroll
  for (int i = 0; i < 8; ++i) { a[i] = lo[i]; a[i + 8] = hi[i]; }
  return a;
}

// ---------------------------------------------------------------------
// Dual-M-tile dense layer: {out0,out1}(16x128) = relu({in0,in1}(16xK) @ W + b).
// Both tiles share each B fragment load -> 1 LDS B load per 2 WMMAs; the two
// WMMAs are independent (distinct D), so they issue back-to-back with no
// hazard NOPs. All A reads precede all stores -> safe in-place with in-order LDS.
// ---------------------------------------------------------------------
template<int KSTEPS>
__device__ __forceinline__ void dense_layer2(const _Float16* in0, const _Float16* in1,
                                             const _Float16* w,
                                             const float* biasLds,   // LDS f32
                                             _Float16* out0, _Float16* out1,
                                             int lane)
{
  v16h a0[KSTEPS], a1[KSTEPS];
#pragma unroll
  for (int ks = 0; ks < KSTEPS; ++ks) {
    a0[ks] = load_afrag(in0, lane, ks);
    a1[ks] = load_afrag(in1, lane, ks);
  }

  const int col   = lane & 15;
  const int rbase = (lane & 16) ? 8 : 0;
#pragma unroll
  for (int nt = 0; nt < 8; ++nt) {
    v8f c0 = {}, c1 = {};
#pragma unroll
    for (int ks = 0; ks < KSTEPS; ++ks) {
      v16h b = *(const v16h*)(w + (((nt * KSTEPS + ks) * 32 + lane) << 4));
      c0 = __builtin_amdgcn_wmma_f32_16x16x32_f16(false, a0[ks], false, b,
                                                  (short)0, c0, false, false);
      c1 = __builtin_amdgcn_wmma_f32_16x16x32_f16(false, a1[ks], false, b,
                                                  (short)0, c1, false, false);
    }
    const int n  = nt * 16 + col;
    const float bv = biasLds[n];
#pragma unroll
    for (int r = 0; r < 8; ++r) {
      float v0 = c0[r] + bv;
      float v1 = c1[r] + bv;
      out0[(r + rbase) * DC + n] = (_Float16)(v0 > 0.f ? v0 : 0.f);
      out1[(r + rbase) * DC + n] = (_Float16)(v1 > 0.f ? v1 : 0.f);
    }
  }
}

// Dual-M-tile head: (16x128) @ (128x16-padded), shared B fragments.
__device__ __forceinline__ void head2(const _Float16* in0, const _Float16* in1,
                                      const _Float16* w, int lane,
                                      v8f& o0, v8f& o1)
{
  v8f c0 = {}, c1 = {};
#pragma unroll
  for (int ks = 0; ks < 4; ++ks) {
    v16h a0 = load_afrag(in0, lane, ks);
    v16h a1 = load_afrag(in1, lane, ks);
    v16h b  = *(const v16h*)(w + ((ks * 32 + lane) << 4));
    c0 = __builtin_amdgcn_wmma_f32_16x16x32_f16(false, a0, false, b,
                                                (short)0, c0, false, false);
    c1 = __builtin_amdgcn_wmma_f32_16x16x32_f16(false, a1, false, b,
                                                (short)0, c1, false, false);
  }
  o0 = c0; o1 = c1;
}

// One full MLP (layer0 + NHID hidden + head) for this wave's 32 points.
__device__ __forceinline__ void mlp_phase(const _Float16* smem, _Float16* myact,
                                          int headw, int outoff,
                                          float* __restrict__ out,
                                          int pwave, int lane, int wave)
{
  const _Float16* wbuf  = smem;
  const _Float16* featb = smem + OFF_FEAT;
  const float*    bl    = (const float*)(smem + OFF_BIAS);
  const int mrow = lane & 15;

  const _Float16* fr0 = featb + (size_t)(wave * 32 + mrow) * 64;
  const _Float16* fr1 = fr0 + 16 * 64;
  _Float16* o0 = myact;
  _Float16* o1 = myact + 16 * DC;

  dense_layer2<2>(fr0, fr1, wbuf + OFF_W0, bl, o0, o1, lane);
  asm volatile("s_wait_dscnt 0" ::: "memory");   // cross-lane LDS RAW fence

  const _Float16* ar0 = myact + mrow * DC;
  const _Float16* ar1 = ar0 + 16 * DC;
#pragma unroll 1
  for (int l = 0; l < NHID; ++l) {
    dense_layer2<4>(ar0, ar1, wbuf + OFF_WH + l * DC * DC, bl + DC + l * DC,
                    o0, o1, lane);
    asm volatile("s_wait_dscnt 0" ::: "memory");
  }

  v8f h0, h1;
  head2(ar0, ar1, wbuf + OFF_HEAD, lane, h0, h1);

  const int col   = lane & 15;
  const int rbase = (lane & 16) ? 8 : 0;
  if (col < headw) {              // EXEC-masked epilogue (after all WMMAs)
    const float bb = bl[4 * DC + col];
#pragma unroll
    for (int r = 0; r < 8; ++r) {
      out[(size_t)(pwave + rbase + r) * 4 + outoff + col]      = h0[r] + bb;
      out[(size_t)(pwave + 16 + rbase + r) * 4 + outoff + col] = h1[r] + bb;
    }
  }
}

__global__ __launch_bounds__(TPB, 1)
void triplane_mlp_kernel(const float* __restrict__ points,
                         const float* __restrict__ planes,
                         const float* __restrict__ dw0, const float* __restrict__ db0,
                         const float* __restrict__ dwh, const float* __restrict__ dbh,
                         const float* __restrict__ aw,  const float* __restrict__ ab,
                         const float* __restrict__ rw0, const float* __restrict__ rb0,
                         const float* __restrict__ rwh, const float* __restrict__ rbh,
                         const float* __restrict__ cw,  const float* __restrict__ cb,
                         float* __restrict__ out)
{
  extern __shared__ __align__(32) _Float16 smem[];
  const int tid   = threadIdx.x;
  const int lane  = tid & 31;
  const int wave  = tid >> 5;
  const int pbase = blockIdx.x * TILE_PTS;

  // Phase 0: per-thread tri-plane sampling + stage density weights/biases
  sample_feat(points, planes, smem + OFF_FEAT, tid, pbase);
  stage_weights(smem, dw0, dwh, aw, /*hN=*/1, db0, dbh, ab, tid);
  __syncthreads();

  _Float16* myact = smem + OFF_ACT + (size_t)wave * 32 * DC;

  // Phase 1: density MLP -> alpha (out[:,3])
  mlp_phase(smem, myact, /*headw=*/1, /*outoff=*/3, out, pbase + wave * 32, lane, wave);
  __syncthreads();   // everyone done with density weights

  // Phase 2: restage rgb weights into same LDS region, run rgb MLP -> out[:,0:3]
  stage_weights(smem, rw0, rwh, cw, /*hN=*/3, rb0, rbh, cb, tid);
  __syncthreads();
  mlp_phase(smem, myact, /*headw=*/3, /*outoff=*/0, out, pbase + wave * 32, lane, wave);
}

extern "C" void kernel_launch(void* const* d_in, const int* in_sizes, int n_in,
                              void* d_out, int out_size, void* d_ws, size_t ws_size,
                              hipStream_t stream) {
  (void)n_in; (void)d_ws; (void)ws_size; (void)out_size;
  const float* points = (const float*)d_in[0];
  const float* planes = (const float*)d_in[1];
  const float* dw0 = (const float*)d_in[2];
  const float* db0 = (const float*)d_in[3];
  const float* dwh = (const float*)d_in[4];
  const float* dbh = (const float*)d_in[5];
  const float* aw  = (const float*)d_in[6];
  const float* ab  = (const float*)d_in[7];
  const float* rw0 = (const float*)d_in[8];
  const float* rb0 = (const float*)d_in[9];
  const float* rwh = (const float*)d_in[10];
  const float* rbh = (const float*)d_in[11];
  const float* cw  = (const float*)d_in[12];
  const float* cb  = (const float*)d_in[13];
  float* out = (float*)d_out;

  const int npts   = in_sizes[0] / 3;
  const int blocks = npts / TILE_PTS;   // 1048576 / 256 = 4096

  hipFuncSetAttribute(reinterpret_cast<const void*>(&triplane_mlp_kernel),
                      hipFuncAttributeMaxDynamicSharedMemorySize, SMEM_BYTES);
  triplane_mlp_kernel<<<blocks, TPB, SMEM_BYTES, stream>>>(
      points, planes, dw0, db0, dwh, dbh, aw, ab,
      rw0, rb0, rwh, rbh, cw, cb, out);
}
